// GPTBigCodeBlock_70119636074917
// MI455X (gfx1250) — matrix operand order repaired and checked by reading, not compiled, not run on hardware
//
#include <hip/hip_runtime.h>
#include <hip/hip_bf16.h>
#include <math.h>

// ---------------------------------------------------------------- types ----
typedef __bf16 bf16_t;
typedef __attribute__((ext_vector_type(16))) __bf16 v16bf;
typedef __attribute__((ext_vector_type(8)))  float  v8f;

union Frag16 {          // one WMMA 16-bit A/B operand: 16 bf16 per lane
    v16bf v;
    uint4 u[2];
    __bf16 e[16];
};

// ------------------------------------------------------------- problem ----
#define Bsz   2
#define Ssz   2048
#define Hsz   4096
#define NHsz  32
#define HDsz  128
#define Isz   16384
#define Msz   (Bsz * Ssz)          // 4096 rows of activations
#define QKVN  (Hsz + 2 * HDsz)     // 4352
#define SCALE_C 0.08838834764831845f
#define LNEPS 1e-5f

__device__ __forceinline__ float gelu_t(float x) {
    float x3 = x * x * x;
    return 0.5f * x * (1.0f + tanhf(0.7978845608028654f * (x + 0.044715f * x3)));
}

// async copy 16B global -> LDS, tracked by ASYNCcnt (CDNA5 path)
__device__ __forceinline__ void async_copy_b128(const bf16_t* g, const bf16_t* l) {
    unsigned lo = (unsigned)(uintptr_t)l;          // LDS address from low 32 bits of pointer
    asm volatile("global_load_async_to_lds_b128 %0, %1, off"
                 :: "v"(lo), "v"(g) : "memory");
}
__device__ __forceinline__ void wait_async0() {
    asm volatile("s_wait_asynccnt 0x0" ::: "memory");
}

// ------------------------------------------------- weight fp32 -> bf16^T ---
__global__ __launch_bounds__(256)
void transpose_cast_bf16(const float* __restrict__ W, bf16_t* __restrict__ Wt,
                         int K, int N) {
    size_t idx = (size_t)blockIdx.x * 256 + threadIdx.x;
    size_t total = (size_t)K * N;
    if (idx >= total) return;
    int k = (int)(idx / N);
    int n = (int)(idx % N);
    Wt[(size_t)n * K + k] = (bf16_t)W[idx];
}

// -------------------------------------------------------------- layernorm --
__global__ __launch_bounds__(256)
void layernorm_bf16(const float* __restrict__ x, const float* __restrict__ w,
                    const float* __restrict__ b, bf16_t* __restrict__ y) {
    const int row  = blockIdx.x;
    const int tid  = threadIdx.x;
    const int lane = tid & 31;
    const int wid  = tid >> 5;

    __shared__ float red[2][8];
    __shared__ float stats[2];

    const float* xr = x + (size_t)row * Hsz;
    float vals[16];
    float s1 = 0.f, s2 = 0.f;
#pragma unroll
    for (int i = 0; i < 16; ++i) {
        float v = xr[tid + i * 256];
        vals[i] = v;
        s1 += v;
        s2 += v * v;
    }
#pragma unroll
    for (int off = 16; off; off >>= 1) {
        s1 += __shfl_xor(s1, off);
        s2 += __shfl_xor(s2, off);
    }
    if (lane == 0) { red[0][wid] = s1; red[1][wid] = s2; }
    __syncthreads();
    if (tid == 0) {
        float a = 0.f, c = 0.f;
#pragma unroll
        for (int i = 0; i < 8; ++i) { a += red[0][i]; c += red[1][i]; }
        float mean = a * (1.0f / Hsz);
        float var  = c * (1.0f / Hsz) - mean * mean;
        stats[0] = mean;
        stats[1] = rsqrtf(var + LNEPS);
    }
    __syncthreads();
    const float mean = stats[0], rstd = stats[1];
    bf16_t* yr = y + (size_t)row * Hsz;
#pragma unroll
    for (int i = 0; i < 16; ++i) {
        int c = tid + i * 256;
        yr[c] = (bf16_t)((vals[i] - mean) * rstd * w[c] + b[c]);
    }
}

// ------------------------------------------------------------------ GEMM ---
// C[M,N] = A[M,K](bf16,row-major) * Bt[N,K](bf16, pre-transposed) + bias
// flags: 1 = add fp32 residual, 2 = tanh-GELU, 4 = store bf16 (else fp32)
// Double-buffered via async global->LDS copies (ASYNCcnt pipeline).
#define BKP 40   // padded LDS K-stride (elements); multiple of 8 -> 16B chunks

__global__ __launch_bounds__(256)
void gemm_bf16(const bf16_t* __restrict__ A, const bf16_t* __restrict__ Bt,
               const float* __restrict__ bias, const float* __restrict__ resid,
               float* __restrict__ Cf, bf16_t* __restrict__ Cb,
               int M, int N, int K, int flags) {
    __shared__ __attribute__((aligned(16))) bf16_t As[2][128 * BKP];
    __shared__ __attribute__((aligned(16))) bf16_t Bs[2][128 * BKP];

    const int tid   = threadIdx.x;
    const int lane  = tid & 31;
    const int wid   = tid >> 5;
    const int wm    = wid >> 2;        // 0..1
    const int wn    = wid & 3;         // 0..3
    const int lrow  = lane & 15;
    const int lhalf = lane >> 4;       // 0/1
    const int m0 = blockIdx.y * 128;
    const int n0 = blockIdx.x * 128;

    v8f acc[4][2] = {};

    // issue one 128x32 A-tile + B-tile as async copies into buffer `buf`
    auto issue_tile = [&](int kb, int buf) {
#pragma unroll
        for (int i = 0; i < 2; ++i) {
            int c  = tid * 2 + i;
            int r  = c >> 2;           // 0..127
            int k8 = (c & 3) * 8;      // 0,8,16,24
            const bf16_t* gA = &A[(size_t)(m0 + r) * K + kb + k8];
            const bf16_t* gB = &Bt[(size_t)(n0 + r) * K + kb + k8];
            async_copy_b128(gA, &As[buf][r * BKP + k8]);
            async_copy_b128(gB, &Bs[buf][r * BKP + k8]);
            // warm L2 one more tile ahead (global_prefetch_b8)
            __builtin_prefetch(gA + 32, 0, 1);
            __builtin_prefetch(gB + 32, 0, 1);
        }
    };

    issue_tile(0, 0);
    wait_async0();
    __syncthreads();

    for (int kb = 0; kb < K; kb += 32) {
        const int cur = (kb >> 5) & 1;
        if (kb + 32 < K) issue_tile(kb + 32, cur ^ 1);  // overlap with WMMA

        Frag16 af[4];
#pragma unroll
        for (int mt = 0; mt < 4; ++mt) {
            int r = wm * 64 + mt * 16 + lrow;
            const bf16_t* p = &As[cur][r * BKP + lhalf * 8];
            af[mt].u[0] = *(const uint4*)p;
            af[mt].u[1] = *(const uint4*)(p + 16);
        }
#pragma unroll
        for (int nt = 0; nt < 2; ++nt) {
            Frag16 bfg;
            int n = wn * 32 + nt * 16 + lrow;
            const bf16_t* p = &Bs[cur][n * BKP + lhalf * 8];
            bfg.u[0] = *(const uint4*)p;
            bfg.u[1] = *(const uint4*)(p + 16);
#pragma unroll
            for (int mt = 0; mt < 4; ++mt)
                acc[mt][nt] = __builtin_amdgcn_wmma_f32_16x16x32_bf16(
                    false, af[mt].v, false, bfg.v, (short)0, acc[mt][nt],
                    false, false);
        }

        wait_async0();        // next tile landed in LDS
        __syncthreads();      // all waves done reading `cur`
    }

    // fused epilogue
#pragma unroll
    for (int mt = 0; mt < 4; ++mt) {
#pragma unroll
        for (int nt = 0; nt < 2; ++nt) {
            int col = n0 + wn * 32 + nt * 16 + lrow;
            float bv = bias[col];
#pragma unroll
            for (int r = 0; r < 8; ++r) {
                int row = m0 + wm * 64 + mt * 16 + lhalf * 8 + r;
                size_t idx = (size_t)row * N + col;
                float v = acc[mt][nt][r] + bv;
                if (flags & 1) v += resid[idx];
                if (flags & 2) v = gelu_t(v);
                if (flags & 4) Cb[idx] = (bf16_t)v;
                else           Cf[idx] = v;
            }
        }
    }
}

// ------------------------------------------------- flash MQA attention ----
// qkv: [Msz, QKVN] bf16  (q: [:,0:4096], k: [:,4096:4224], v: [:,4224:4352])
// attn: [Msz, Hsz] bf16  (col = head*HD + d)
__global__ __launch_bounds__(256)
void mqa_attention(const bf16_t* __restrict__ qkv, bf16_t* __restrict__ attn) {
    const int qblk = blockIdx.x;     // 16 blocks of 128 q-rows
    const int head = blockIdx.y;
    const int bidx = blockIdx.z;
    const int tid   = threadIdx.x;
    const int lane  = tid & 31;
    const int wid   = tid >> 5;      // 8 waves, 16 q-rows each
    const int lrow  = lane & 15;
    const int lhalf = lane >> 4;

    __shared__ __attribute__((aligned(16))) bf16_t Vt[HDsz * 40]; // [d][kv]
    __shared__ __attribute__((aligned(16))) bf16_t Pj[8][16 * 32];

    const size_t RS   = QKVN;
    const size_t base = (size_t)bidx * Ssz;

    // Q fragments for this wave's 16 rows (A layout: 8+8 K split per half)
    Frag16 qf[4];
    {
        int qrow = qblk * 128 + wid * 16 + lrow;
        const bf16_t* qp = qkv + (base + qrow) * RS + head * HDsz;
#pragma unroll
        for (int kk = 0; kk < 4; ++kk) {
            const bf16_t* p = qp + kk * 32 + lhalf * 8;
            qf[kk].u[0] = *(const uint4*)p;
            qf[kk].u[1] = *(const uint4*)(p + 16);
        }
    }

    v8f o[8] = {};
    float mrun[8], lrun[8];
#pragma unroll
    for (int r = 0; r < 8; ++r) { mrun[r] = -1e30f; lrun[r] = 0.f; }

    const int jtiles = qblk * 4 + 4;            // causal: kv <= q-block end
    for (int jt = 0; jt < jtiles; ++jt) {
        __syncthreads();                        // Vt reuse barrier
        {   // stage V tile (32 kv x 128 d) transposed into Vt[d][kv]
            int r     = tid >> 3;               // kv row 0..31
            int dbase = (tid & 7) * 16;
            Frag16 tmp;
            const bf16_t* vp =
                qkv + (base + jt * 32 + r) * RS + Hsz + HDsz + dbase;
            tmp.u[0] = *(const uint4*)vp;
            tmp.u[1] = *(const uint4*)(vp + 8);
#pragma unroll
            for (int e = 0; e < 16; ++e) Vt[(dbase + e) * 40 + r] = tmp.e[e];
        }
        __syncthreads();

        // scores S = Q K^T : B-frags straight from global (shared KV head)
        v8f sc[2] = {};
#pragma unroll
        for (int nt = 0; nt < 2; ++nt) {
            int kvn = jt * 32 + nt * 16 + lrow;
            const bf16_t* kp = qkv + (base + kvn) * RS + Hsz;
#pragma unroll
            for (int kk = 0; kk < 4; ++kk) {
                Frag16 kf;
                const bf16_t* p = kp + kk * 32 + lhalf * 8;
                kf.u[0] = *(const uint4*)p;
                kf.u[1] = *(const uint4*)(p + 16);
                sc[nt] = __builtin_amdgcn_wmma_f32_16x16x32_bf16(
                    false, qf[kk].v, false, kf.v, (short)0, sc[nt],
                    false, false);
            }
        }

        // scale + causal mask + online softmax (rows = vgpr + 8*lhalf)
        const int qg = qblk * 128 + wid * 16 + lhalf * 8;
#pragma unroll
        for (int r = 0; r < 8; ++r) {
            int qrowg = qg + r;
            int kv0   = jt * 32 + lrow;
            float s0 = sc[0][r] * SCALE_C;
            float s1 = sc[1][r] * SCALE_C;
            if (kv0 > qrowg)      s0 = -1e30f;
            if (kv0 + 16 > qrowg) s1 = -1e30f;
            float mx = fmaxf(s0, s1);
#pragma unroll
            for (int off = 8; off; off >>= 1)
                mx = fmaxf(mx, __shfl_xor(mx, off));   // 16-lane row reduce
            float mnew  = fmaxf(mrun[r], mx);
            float alpha = __expf(mrun[r] - mnew);
            float p0 = __expf(s0 - mnew);
            float p1 = __expf(s1 - mnew);
            sc[0][r] = p0;
            sc[1][r] = p1;
            float rowsum = p0 + p1;
#pragma unroll
            for (int off = 8; off; off >>= 1)
                rowsum += __shfl_xor(rowsum, off);
            lrun[r] = lrun[r] * alpha + rowsum;
            mrun[r] = mnew;
#pragma unroll
            for (int t = 0; t < 8; ++t) o[t][r] *= alpha;
        }

        // P (C layout) -> bf16 A layout via per-wave LDS patch
        bf16_t* pw = Pj[wid];
#pragma unroll
        for (int nt = 0; nt < 2; ++nt)
#pragma unroll
            for (int r = 0; r < 8; ++r)
                pw[(lhalf * 8 + r) * 32 + nt * 16 + lrow] = (bf16_t)sc[nt][r];
        __syncthreads();

        Frag16 pf;
        {
            const bf16_t* p = pw + lrow * 32 + lhalf * 8;
            pf.u[0] = *(const uint4*)p;
            pf.u[1] = *(const uint4*)(p + 16);
        }
        // O += P V : V B-frags contiguous from transposed LDS tile
#pragma unroll
        for (int t = 0; t < 8; ++t) {
            Frag16 vf;
            const bf16_t* p = &Vt[(t * 16 + lrow) * 40 + lhalf * 8];
            vf.u[0] = *(const uint4*)p;
            vf.u[1] = *(const uint4*)(p + 16);
            o[t] = __builtin_amdgcn_wmma_f32_16x16x32_bf16(
                false, pf.v, false, vf.v, (short)0, o[t], false, false);
        }
    }

    // epilogue: O /= l, store bf16
#pragma unroll
    for (int r = 0; r < 8; ++r) {
        float inv = 1.0f / lrun[r];
        int row = qblk * 128 + wid * 16 + lhalf * 8 + r;
        bf16_t* op = attn + (base + row) * Hsz + head * HDsz;
#pragma unroll
        for (int t = 0; t < 8; ++t)
            op[t * 16 + lrow] = (bf16_t)(o[t][r] * inv);
    }
}

// ---------------------------------------------------------------- launch ---
extern "C" void kernel_launch(void* const* d_in, const int* in_sizes, int n_in,
                              void* d_out, int out_size, void* d_ws,
                              size_t ws_size, hipStream_t stream) {
    const float* x           = (const float*)d_in[0];
    const float* ln1_w       = (const float*)d_in[1];
    const float* ln1_b       = (const float*)d_in[2];
    const float* w_qkv       = (const float*)d_in[3];
    const float* b_qkv       = (const float*)d_in[4];
    const float* w_attn_proj = (const float*)d_in[5];
    const float* b_attn_proj = (const float*)d_in[6];
    const float* ln2_w       = (const float*)d_in[7];
    const float* ln2_b       = (const float*)d_in[8];
    const float* w_fc        = (const float*)d_in[9];
    const float* b_fc        = (const float*)d_in[10];
    const float* w_mlp_proj  = (const float*)d_in[11];
    const float* b_mlp_proj  = (const float*)d_in[12];
    float* out = (float*)d_out;

    char* ws = (char*)d_ws;
    size_t off = 0;
    auto carve = [&](size_t bytes) -> void* {
        void* p = ws + off;
        off += bytes;
        off = (off + 255) & ~(size_t)255;
        return p;
    };
    bf16_t* wqkvT = (bf16_t*)carve((size_t)QKVN * Hsz * 2);
    bf16_t* wattT = (bf16_t*)carve((size_t)Hsz * Hsz * 2);
    bf16_t* wfcT  = (bf16_t*)carve((size_t)Isz * Hsz * 2);
    bf16_t* wmlpT = (bf16_t*)carve((size_t)Hsz * Isz * 2);
    bf16_t* ln1o  = (bf16_t*)carve((size_t)Msz * Hsz * 2);
    bf16_t* qkvb  = (bf16_t*)carve((size_t)Msz * QKVN * 2);
    bf16_t* atno  = (bf16_t*)carve((size_t)Msz * Hsz * 2);
    float*  h1    = (float*) carve((size_t)Msz * Hsz * 4);
    bf16_t* ln2o  = (bf16_t*)carve((size_t)Msz * Hsz * 2);
    bf16_t* fca   = (bf16_t*)carve((size_t)Msz * Isz * 2);
    (void)ws_size; (void)in_sizes; (void)n_in; (void)out_size;

    auto blks = [](size_t total) { return (unsigned)((total + 255) / 256); };

    // 1) weights -> bf16, pre-transposed [N,K]
    transpose_cast_bf16<<<blks((size_t)Hsz * QKVN), 256, 0, stream>>>(
        w_qkv, wqkvT, Hsz, QKVN);
    transpose_cast_bf16<<<blks((size_t)Hsz * Hsz), 256, 0, stream>>>(
        w_attn_proj, wattT, Hsz, Hsz);
    transpose_cast_bf16<<<blks((size_t)Hsz * Isz), 256, 0, stream>>>(
        w_fc, wfcT, Hsz, Isz);
    transpose_cast_bf16<<<blks((size_t)Isz * Hsz), 256, 0, stream>>>(
        w_mlp_proj, wmlpT, Isz, Hsz);

    // 2) LN1
    layernorm_bf16<<<Msz, 256, 0, stream>>>(x, ln1_w, ln1_b, ln1o);

    // 3) QKV projection -> bf16 (+bias)
    gemm_bf16<<<dim3(QKVN / 128, Msz / 128), 256, 0, stream>>>(
        ln1o, wqkvT, b_qkv, nullptr, nullptr, qkvb, Msz, QKVN, Hsz, /*flags*/4);

    // 4) flash MQA attention
    mqa_attention<<<dim3(Ssz / 128, NHsz, Bsz), 256, 0, stream>>>(qkvb, atno);

    // 5) attn proj + bias + residual(x) -> h1 fp32
    gemm_bf16<<<dim3(Hsz / 128, Msz / 128), 256, 0, stream>>>(
        atno, wattT, b_attn_proj, x, h1, nullptr, Msz, Hsz, Hsz, /*flags*/1);

    // 6) LN2
    layernorm_bf16<<<Msz, 256, 0, stream>>>(h1, ln2_w, ln2_b, ln2o);

    // 7) FC + bias + GELU -> bf16
    gemm_bf16<<<dim3(Isz / 128, Msz / 128), 256, 0, stream>>>(
        ln2o, wfcT, b_fc, nullptr, nullptr, fca, Msz, Isz, Hsz, /*flags*/4 | 2);

    // 8) MLP proj + bias + residual(h1) -> out fp32
    gemm_bf16<<<dim3(Hsz / 128, Msz / 128), 256, 0, stream>>>(
        fca, wmlpT, b_mlp_proj, h1, out, nullptr, Msz, Hsz, Isz, /*flags*/1);
}
